// HiveValueGNN_19009525252309
// MI455X (gfx1250) — matrix operand order repaired and checked by reading, not compiled, not run on hardware
//
#include <hip/hip_runtime.h>
#include <hip/hip_bf16.h>

typedef __attribute__((ext_vector_type(16))) _Float16 v16h;
typedef __attribute__((ext_vector_type(8)))  _Float16 v8h;
typedef __attribute__((ext_vector_type(8)))  float    v8f;

#define N_NODES  50000
#define N_EDGES  800000
#define FDIM     128
#define N_REL    8
#define N_GRAPHS 64

// ---------------------------------------------------------------- utilities

__global__ void zero_f32_kernel(float* __restrict__ p, int n) {
  int i = blockIdx.x * blockDim.x + threadIdx.x;
  if (i < n) p[i] = 0.0f;
}

__global__ void count_edges_kernel(const int* __restrict__ etype,
                                   const int* __restrict__ dst,
                                   float* __restrict__ cnt) {
  int e = blockIdx.x * blockDim.x + threadIdx.x;
  if (e < N_EDGES) atomicAdd(&cnt[etype[e] * N_NODES + dst[e]], 1.0f);
}

__global__ void inv_counts_kernel(float* __restrict__ cnt, int n) {
  int i = blockIdx.x * blockDim.x + threadIdx.x;
  if (i < n) cnt[i] = 1.0f / fmaxf(cnt[i], 1.0f);
}

__global__ void relu_ip_kernel(float* __restrict__ p, int n) {
  int i = blockIdx.x * blockDim.x + threadIdx.x;
  if (i < n) p[i] = fmaxf(p[i], 0.0f);
}

// f32 -> f16 feature encode (done once per layer, reused by 9 GEMM passes)
__global__ void encode_f16_kernel(const float* __restrict__ in,
                                  _Float16* __restrict__ out, int n) {
  int i = blockIdx.x * blockDim.x + threadIdx.x;
  if (i < n) out[i] = (_Float16)in[i];
}

// Weight encode: nmat matrices [128 x 128] row-major f32 (in[m][k][col])
// -> transposed f16 (out[m][col][k]) so each WMMA B-lane reads contiguous K.
__global__ void encode_wT_kernel(const float* __restrict__ in,
                                 _Float16* __restrict__ out, int nmat) {
  int i = blockIdx.x * blockDim.x + threadIdx.x;
  int total = nmat * FDIM * FDIM;
  if (i >= total) return;
  int m   = i / (FDIM * FDIM);
  int rem = i - m * (FDIM * FDIM);
  int col = rem / FDIM;
  int k   = rem - col * FDIM;
  out[i] = (_Float16)in[m * (FDIM * FDIM) + k * FDIM + col];
}

// ------------------------------------------------- WMMA GEMM: [N,128]x[128,128]
// One wave computes one 16x16 f32 tile; block = (32,8) covers all 8 column
// tiles of one row tile. A tile (16x128 f16 = 4 KB) staged in LDS once per
// block; per K-step: 2x ds_load_b128 (A) + 2x global_load_b128 (B) + 1 WMMA.

__global__ __launch_bounds__(256)
void wmma_gemm_f16_kernel(const _Float16* __restrict__ Ah,   // [N,128] row-major
                          const _Float16* __restrict__ Bt,   // [128,128] col-major
                          const float* __restrict__ bias,
                          float* __restrict__ Out) {
  __shared__ _Float16 atile[16 * FDIM];      // 4 KB

  const int lane   = threadIdx.x;            // 0..31
  const int tile_n = threadIdx.y;            // 0..7
  const int tile_m = blockIdx.x;             // 0..3124 (50000/16 exact)
  const int l15    = lane & 15;
  const int hi     = lane >> 4;
  const int tid    = tile_n * 32 + lane;     // 0..255

  // cooperative A-tile load: 256 threads x 16 B = 4 KB contiguous
  {
    const v8h* __restrict__ srcv =
        reinterpret_cast<const v8h*>(Ah + (size_t)tile_m * 16 * FDIM);
    v8h* dstv = reinterpret_cast<v8h*>(atile);
    dstv[tid] = srcv[tid];
  }
  __syncthreads();

  const int col = tile_n * 16 + l15;
  const _Float16* __restrict__ arow = atile + l15 * FDIM;
  const _Float16* __restrict__ brow = Bt + (size_t)col * FDIM;

  v8f c = {};
  #pragma unroll
  for (int k0 = 0; k0 < FDIM; k0 += 32) {
    // A 16x32: lanes 0-15 hold K {k0..+7, k0+16..+23}; lanes 16-31 shift by 8
    const int ka = k0 + hi * 8;
    v8h a0 = *reinterpret_cast<const v8h*>(arow + ka);
    v8h a1 = *reinterpret_cast<const v8h*>(arow + ka + 16);
    v16h a = __builtin_shufflevector(a0, a1, 0, 1, 2, 3, 4, 5, 6, 7,
                                             8, 9, 10, 11, 12, 13, 14, 15);
    // B 32x16: lane = column; K {k0..+7, k0+16..+23}; lanes 16-31 shift by 8
    const int kb = k0 + hi * 8;
    v8h b0 = *reinterpret_cast<const v8h*>(brow + kb);
    v8h b1 = *reinterpret_cast<const v8h*>(brow + kb + 16);
    v16h b = __builtin_shufflevector(b0, b1, 0, 1, 2, 3, 4, 5, 6, 7,
                                             8, 9, 10, 11, 12, 13, 14, 15);

    c = __builtin_amdgcn_wmma_f32_16x16x32_f16(
            /*neg_a=*/false, a, /*neg_b=*/false, b,
            /*c_mod=*/(short)0, c, /*reuse_a=*/false, /*reuse_b=*/false);
  }

  // C/D layout: lanes 0-15 VGPR j -> M=j; lanes 16-31 VGPR j -> M=8+j
  const float bv = bias ? bias[col] : 0.0f;
  #pragma unroll
  for (int j = 0; j < 8; ++j) {
    const int orow = tile_m * 16 + hi * 8 + j;
    Out[(size_t)orow * FDIM + col] = c[j] + bv;
  }
}

// ------------------------------------- scaled scatter: out[dst] += T[src]*inv
// One wave per edge; lane covers 4 coalesced columns.

__global__ void scatter_rel_kernel(const float* __restrict__ T,
                                   const int* __restrict__ src,
                                   const int* __restrict__ dst,
                                   const int* __restrict__ etype,
                                   const float* __restrict__ cinv_r,
                                   int rel,
                                   float* __restrict__ Hout) {
  int idx  = blockIdx.x * blockDim.x + threadIdx.x;
  int e    = idx >> 5;
  int lane = idx & 31;
  if (e >= N_EDGES) return;
  if (etype[e] != rel) return;
  int s = src[e], d = dst[e];
  float sc = cinv_r[d];
  const float* __restrict__ ts = T + (size_t)s * FDIM;
  float* __restrict__ hd = Hout + (size_t)d * FDIM;
  #pragma unroll
  for (int q = 0; q < 4; ++q) {
    int cix = lane + q * 32;
    atomicAdd(&hd[cix], ts[cix] * sc);
  }
}

// -------------------------------------------------- pooling: g[batch[n]] += h[n]

__global__ void pool_kernel(const float* __restrict__ h,
                            const int* __restrict__ batch,
                            float* __restrict__ g) {
  int idx = blockIdx.x * blockDim.x + threadIdx.x;
  if (idx >= N_NODES * FDIM) return;
  int n = idx / FDIM;
  int cix = idx - n * FDIM;
  atomicAdd(&g[batch[n] * FDIM + cix], h[idx]);
}

// ------------------------------------ final MLP: tanh(relu(g@mw1+mb1)@mw2+mb2)

__global__ __launch_bounds__(256)
void mlp_kernel(const float* __restrict__ g,
                const float* __restrict__ mw1, const float* __restrict__ mb1,
                const float* __restrict__ mw2, const float* __restrict__ mb2,
                float* __restrict__ out) {
  __shared__ float hid[N_GRAPHS * (FDIM / 2)];
  int tid = threadIdx.x;
  for (int idx = tid; idx < N_GRAPHS * (FDIM / 2); idx += 256) {
    int i = idx / (FDIM / 2);
    int j = idx - i * (FDIM / 2);
    float s = mb1[j];
    #pragma unroll 4
    for (int k = 0; k < FDIM; ++k)
      s += g[i * FDIM + k] * mw1[k * (FDIM / 2) + j];
    hid[idx] = fmaxf(s, 0.0f);
  }
  __syncthreads();
  for (int i = tid; i < N_GRAPHS; i += 256) {
    float s = mb2[0];
    #pragma unroll 4
    for (int j = 0; j < FDIM / 2; ++j)
      s += hid[i * (FDIM / 2) + j] * mw2[j];
    out[i] = tanhf(s);
  }
}

// ---------------------------------------------------------------- launcher

extern "C" void kernel_launch(void* const* d_in, const int* in_sizes, int n_in,
                              void* d_out, int out_size, void* d_ws, size_t ws_size,
                              hipStream_t stream) {
  (void)in_sizes; (void)n_in; (void)out_size; (void)ws_size;

  const float* x     = (const float*)d_in[0];
  const int*   ei    = (const int*)  d_in[1];
  const int*   srcp  = ei;
  const int*   dstp  = ei + N_EDGES;
  const int*   et    = (const int*)  d_in[2];
  const int*   batch = (const int*)  d_in[3];
  const float* W1    = (const float*)d_in[4];
  const float* root1 = (const float*)d_in[5];
  const float* b1    = (const float*)d_in[6];
  const float* W2    = (const float*)d_in[7];
  const float* root2 = (const float*)d_in[8];
  const float* b2    = (const float*)d_in[9];
  const float* mw1   = (const float*)d_in[10];
  const float* mb1   = (const float*)d_in[11];
  const float* mw2   = (const float*)d_in[12];
  const float* mb2   = (const float*)d_in[13];
  float* out = (float*)d_out;

  const int NF  = N_NODES * FDIM;
  const int NRN = N_REL * N_NODES;
  const int WM  = FDIM * FDIM;              // one weight matrix

  char* ws = (char*)d_ws;
  float*     T    = (float*)ws;     ws += (size_t)NF * sizeof(float);
  float*     h1   = (float*)ws;     ws += (size_t)NF * sizeof(float);
  float*     h2   = (float*)ws;     ws += (size_t)NF * sizeof(float);
  float*     cinv = (float*)ws;     ws += (size_t)NRN * sizeof(float);
  float*     g    = (float*)ws;     ws += (size_t)N_GRAPHS * FDIM * sizeof(float);
  _Float16*  Xh   = (_Float16*)ws;  ws += (size_t)NF * sizeof(_Float16);
  _Float16*  Wt   = (_Float16*)ws;  ws += (size_t)(N_REL + 1) * WM * sizeof(_Float16);

  // per-(relation,dst) mean denominators
  zero_f32_kernel<<<(NRN + 255) / 256, 256, 0, stream>>>(cinv, NRN);
  count_edges_kernel<<<(N_EDGES + 255) / 256, 256, 0, stream>>>(et, dstp, cinv);
  inv_counts_kernel<<<(NRN + 255) / 256, 256, 0, stream>>>(cinv, NRN);

  const dim3 gemm_grid(N_NODES / 16, 1, 1);
  const dim3 gemm_block(32, 8, 1);
  const int scat_blocks = (N_EDGES * 32 + 255) / 256;
  const int enc_blocks  = (NF + 255) / 256;

  // ---- layer 1: h1 = relu(x@root1 + b1 + sum_r mean_r(x@W1_r))
  encode_f16_kernel<<<enc_blocks, 256, 0, stream>>>(x, Xh, NF);
  encode_wT_kernel<<<(N_REL * WM + 255) / 256, 256, 0, stream>>>(W1, Wt, N_REL);
  encode_wT_kernel<<<(WM + 255) / 256, 256, 0, stream>>>(root1, Wt + (size_t)N_REL * WM, 1);

  wmma_gemm_f16_kernel<<<gemm_grid, gemm_block, 0, stream>>>(
      Xh, Wt + (size_t)N_REL * WM, b1, h1);
  for (int r = 0; r < N_REL; ++r) {
    wmma_gemm_f16_kernel<<<gemm_grid, gemm_block, 0, stream>>>(
        Xh, Wt + (size_t)r * WM, nullptr, T);
    scatter_rel_kernel<<<scat_blocks, 256, 0, stream>>>(
        T, srcp, dstp, et, cinv + (size_t)r * N_NODES, r, h1);
  }
  relu_ip_kernel<<<enc_blocks, 256, 0, stream>>>(h1, NF);

  // ---- layer 2: h2 = relu(h1@root2 + b2 + sum_r mean_r(h1@W2_r))
  encode_f16_kernel<<<enc_blocks, 256, 0, stream>>>(h1, Xh, NF);
  encode_wT_kernel<<<(N_REL * WM + 255) / 256, 256, 0, stream>>>(W2, Wt, N_REL);
  encode_wT_kernel<<<(WM + 255) / 256, 256, 0, stream>>>(root2, Wt + (size_t)N_REL * WM, 1);

  wmma_gemm_f16_kernel<<<gemm_grid, gemm_block, 0, stream>>>(
      Xh, Wt + (size_t)N_REL * WM, b2, h2);
  for (int r = 0; r < N_REL; ++r) {
    wmma_gemm_f16_kernel<<<gemm_grid, gemm_block, 0, stream>>>(
        Xh, Wt + (size_t)r * WM, nullptr, T);
    scatter_rel_kernel<<<scat_blocks, 256, 0, stream>>>(
        T, srcp, dstp, et, cinv + (size_t)r * N_NODES, r, h2);
  }
  relu_ip_kernel<<<enc_blocks, 256, 0, stream>>>(h2, NF);

  // ---- pool + value MLP
  zero_f32_kernel<<<(N_GRAPHS * FDIM + 255) / 256, 256, 0, stream>>>(g, N_GRAPHS * FDIM);
  pool_kernel<<<(NF + 255) / 256, 256, 0, stream>>>(h2, batch, g);
  mlp_kernel<<<1, 256, 0, stream>>>(g, mw1, mb1, mw2, mb2, out);
}
